// RelationUpdateSubject_20005957665470
// MI455X (gfx1250) — matrix operand order repaired and checked by reading, not compiled, not run on hardware
//
#include <hip/hip_runtime.h>
#include <hip/hip_bf16.h>

typedef __attribute__((ext_vector_type(16))) _Float16 v16h;
typedef __attribute__((ext_vector_type(8)))  _Float16 v8h;
typedef __attribute__((ext_vector_type(4)))  _Float16 v4h;
typedef __attribute__((ext_vector_type(8)))  float    v8f;

#define D_DIM 128
#define K_DIM 256
#define PITCH 264            // f16 elements per LDS row: 256 + 8 pad -> conflict-free b128 loads
#define ROWS_PER_BLOCK 80    // 5 M-tiles of 16 rows
#define THREADS 320          // 10 waves: (tile, N-half) pairs

__device__ __forceinline__ v8f wmma_f16(v16h a, v16h b, v8f c) {
    // D = A(16x32 f16) x B(32x16 f16) + C(16x16 f32)
    return __builtin_amdgcn_wmma_f32_16x16x32_f16(
        /*neg_a=*/false, a, /*neg_b=*/false, b,
        /*c_mod=*/(short)0, c, /*reuse_a=*/false, /*reuse_b=*/false);
}

__global__ __launch_bounds__(THREADS)
void RelationUpdateSubject_main_kernel(
    const float* __restrict__ S, const float* __restrict__ O,
    const float* __restrict__ Vs, const float* __restrict__ Vo,
    const float* __restrict__ Ws, const float* __restrict__ Wo,
    const float* __restrict__ W, float* __restrict__ Out)
{
    __shared__ __attribute__((aligned(16))) _Float16 lds_bg[D_DIM * PITCH]; // [n][k] gate weights f16
    __shared__ __attribute__((aligned(16))) _Float16 lds_bd[D_DIM * PITCH]; // [n][k] dir  weights f16
    __shared__ float lds_dot[ROWS_PER_BLOCK];

    const int tid  = threadIdx.x;
    const int lane = tid & 31;
    const int wid  = tid >> 5;

    // ---------------- stage weights in LDS as f16 (k-contiguous, per-n rows) ----
    // B_gate[n][k] = k<128 ? Vs[n][k] : Vo[n][k-128]; same for dir with Ws/Wo.
    {
        const int NGROUPS = D_DIM * (K_DIM / 4); // groups of 4 k
        for (int g = tid; g < NGROUPS; g += THREADS) {
            int n  = g >> 6;          // 64 groups per n-row
            int kg = (g & 63) << 2;   // k start (never straddles k=128)
            const float* sg = (kg < 128) ? (Vs + n * 128 + kg) : (Vo + n * 128 + kg - 128);
            const float* sd = (kg < 128) ? (Ws + n * 128 + kg) : (Wo + n * 128 + kg - 128);
            float4 ag = *(const float4*)sg;
            float4 ad = *(const float4*)sd;
            v4h hg, hd;
            hg[0] = (_Float16)ag.x; hg[1] = (_Float16)ag.y; hg[2] = (_Float16)ag.z; hg[3] = (_Float16)ag.w;
            hd[0] = (_Float16)ad.x; hd[1] = (_Float16)ad.y; hd[2] = (_Float16)ad.z; hd[3] = (_Float16)ad.w;
            *(v4h*)&lds_bg[n * PITCH + kg] = hg;
            *(v4h*)&lds_bd[n * PITCH + kg] = hd;
        }
    }
    // ---------------- per-row dot(o, w) ----------------------------------------
    const int blockRow = blockIdx.x * ROWS_PER_BLOCK;
    if (tid < ROWS_PER_BLOCK) {
        const float* orow = O + (size_t)(blockRow + tid) * D_DIM;
        float acc = 0.f;
        #pragma unroll
        for (int k = 0; k < D_DIM; k += 4) {
            float4 ov = *(const float4*)(orow + k);
            float4 wv = *(const float4*)(W + k);
            acc += ov.x * wv.x + ov.y * wv.y + ov.z * wv.z + ov.w * wv.w;
        }
        lds_dot[tid] = acc;
    }
    __syncthreads();

    // ---------------- wave work assignment -------------------------------------
    const int tilep  = wid >> 1;          // 0..4 : which 16-row M-tile
    const int half   = wid & 1;           // 0..1 : which half of N (4 strips of 16)
    const int ntbase = half * 4;
    const int arow   = blockRow + tilep * 16 + (lane & 15);   // A-operand row for this lane
    const int hi16   = lane >> 4;                              // 0 or 1

    v8f accg[4] = {}, accd[4] = {};

    // ---------------- K loop: A=[s|o] split f32 -> f16_hi + f16_lo --------------
    for (int ks = 0; ks < 8; ++ks) {
        const int k0 = ks * 32;
        const float* Abase = ((k0 < 128) ? S : O) + (size_t)arow * D_DIM + (k0 & 127);
        const int c0 = hi16 * 16; // lanes 0-15: K 0-15; lanes 16-31: K 16-31

        float fa[16];
        {
            float4 t;
            t = *(const float4*)(Abase + c0 +  0); fa[0]=t.x; fa[1]=t.y; fa[2]=t.z;  fa[3]=t.w;
            t = *(const float4*)(Abase + c0 +  4); fa[4]=t.x; fa[5]=t.y; fa[6]=t.z;  fa[7]=t.w;
            t = *(const float4*)(Abase + c0 +  8); fa[8]=t.x; fa[9]=t.y; fa[10]=t.z; fa[11]=t.w;
            t = *(const float4*)(Abase + c0 + 12); fa[12]=t.x; fa[13]=t.y; fa[14]=t.z; fa[15]=t.w;
        }
        v16h ahi, alo;
        #pragma unroll
        for (int i = 0; i < 16; ++i) {
            float x = fa[i];
            _Float16 h = (_Float16)x;
            ahi[i] = h;
            alo[i] = (_Float16)(x - (float)h);
        }

        const int koff = k0 + (hi16 << 4);  // B: lanes 0-15 K=k0..k0+15, lanes 16-31 K=k0+16..+31
        #pragma unroll
        for (int j = 0; j < 4; ++j) {
            const int n = (ntbase + j) * 16 + (lane & 15);
            const _Float16* bg = &lds_bg[n * PITCH + koff];
            const _Float16* bd = &lds_bd[n * PITCH + koff];
            v8h g0 = *(const v8h*)bg, g1 = *(const v8h*)(bg + 8);
            v8h d0 = *(const v8h*)bd, d1 = *(const v8h*)(bd + 8);
            v16h bgv, bdv;
            #pragma unroll
            for (int i = 0; i < 8; ++i) {
                bgv[i] = g0[i]; bgv[8 + i] = g1[i];
                bdv[i] = d0[i]; bdv[8 + i] = d1[i];
            }
            accg[j] = wmma_f16(ahi, bgv, accg[j]);
            accg[j] = wmma_f16(alo, bgv, accg[j]);
            accd[j] = wmma_f16(ahi, bdv, accd[j]);
            accd[j] = wmma_f16(alo, bdv, accd[j]);
        }
    }

    // ---------------- epilogue: interaction + sigmoid*relu ----------------------
    // C/D layout: element e at lane l -> row = mtile_base + e + (l>=16 ? 8:0), col = strip*16 + (l&15)
    const int mb = tilep * 16 + hi16 * 8;   // local row base for this lane's C elements
    float dots[8];
    #pragma unroll
    for (int e = 0; e < 8; ++e) dots[e] = lds_dot[mb + e];

    const size_t rbase = (size_t)(blockRow + mb) * D_DIM;
    #pragma unroll
    for (int j = 0; j < 4; ++j) {
        const int n = (ntbase + j) * 16 + (lane & 15);
        #pragma unroll
        for (int e = 0; e < 8; ++e) {
            size_t off = rbase + (size_t)e * D_DIM + n;
            float sval = S[off];
            float dv = accd[j][e] + sval * dots[e];
            float gv = accg[j][e];
            float gate = 1.f / (1.f + __expf(-gv));
            Out[off] = gate * (dv > 0.f ? dv : 0.f);
        }
    }
}

// Generic scalar tail (only used if rows % 80 != 0; unused for BR = 1e6).
__global__ void RelationUpdateSubject_tail_kernel(
    const float* __restrict__ S, const float* __restrict__ O,
    const float* __restrict__ Vs, const float* __restrict__ Vo,
    const float* __restrict__ Ws, const float* __restrict__ Wo,
    const float* __restrict__ W, float* __restrict__ Out, int rowStart)
{
    int row = rowStart + blockIdx.x;
    int n = threadIdx.x;
    const float* s = S + (size_t)row * D_DIM;
    const float* o = O + (size_t)row * D_DIM;
    float g = 0.f, d = 0.f, dw = 0.f;
    for (int k = 0; k < D_DIM; ++k) {
        float sv = s[k], ov = o[k];
        g += sv * Vs[n * 128 + k] + ov * Vo[n * 128 + k];
        d += sv * Ws[n * 128 + k] + ov * Wo[n * 128 + k];
        dw += ov * W[k];
    }
    d += s[n] * dw;
    float gate = 1.f / (1.f + __expf(-g));
    Out[(size_t)row * D_DIM + n] = gate * fmaxf(d, 0.f);
}

extern "C" void kernel_launch(void* const* d_in, const int* in_sizes, int n_in,
                              void* d_out, int out_size, void* d_ws, size_t ws_size,
                              hipStream_t stream) {
    const float* S  = (const float*)d_in[0];
    const float* O  = (const float*)d_in[1];
    const float* Vs = (const float*)d_in[2];
    const float* Vo = (const float*)d_in[3];
    const float* Ws = (const float*)d_in[4];
    const float* Wo = (const float*)d_in[5];
    const float* W  = (const float*)d_in[6];
    float* Out = (float*)d_out;

    const int BR = in_sizes[0] / D_DIM;
    const int nb = BR / ROWS_PER_BLOCK;
    if (nb > 0) {
        RelationUpdateSubject_main_kernel<<<nb, THREADS, 0, stream>>>(
            S, O, Vs, Vo, Ws, Wo, W, Out);
    }
    const int tail = BR - nb * ROWS_PER_BLOCK;
    if (tail > 0) {
        RelationUpdateSubject_tail_kernel<<<tail, D_DIM, 0, stream>>>(
            S, O, Vs, Vo, Ws, Wo, W, Out, nb * ROWS_PER_BLOCK);
    }
}
